// Self_Attntion_Unit_80839874445870
// MI455X (gfx1250) — matrix operand order repaired and checked by reading, not compiled, not run on hardware
//
#include <hip/hip_runtime.h>

// Self-attention unit, fused flash-attention style, bf16 WMMA (gfx1250 / CDNA5).
// x: [8, 2048, 256] f32; x_mask: [8, 2048] bool (1 byte, True = masked key).
// out[b,q,:] = softmax_k( mask(diagzero(x[b,q]·x[b,k])) ) · x[b,k,:]
//
// Path A (ws_size >= 16.8MB): pre-pass converts x -> bf16 row-major + bf16
//   transposed [b][d][l]; attention kernel double-buffers K/V tiles with
//   async global->LDS b128 copies (ASYNCcnt) overlapped with WMMA compute.
// Path B: convert-in-kernel staging (single buffer).

#define B_SZ   8
#define L_SEQ  2048
#define D_DIM  256
#define Q_TILE 128          // queries per workgroup
#define N_WAVE 8            // 8 waves x 16-query strips
#define K_TILE 64           // keys per inner iteration
#define N_SUB  (K_TILE/16)  // 4 key subtiles of 16

#if __has_builtin(__builtin_amdgcn_global_load_async_to_lds_b128) && \
    __has_builtin(__builtin_amdgcn_s_wait_asynccnt)
#define HAVE_ASYNC_LDS 1
#else
#define HAVE_ASYNC_LDS 0
#endif

typedef __attribute__((ext_vector_type(16))) __bf16 v16bf;
typedef __attribute__((ext_vector_type(8)))  float  v8f;

union Frag {
    v16bf        v;
    unsigned int u[8];
};

// f32 -> bf16 via hardware convert (RNE), bit pattern as ushort
__device__ __forceinline__ unsigned short f2bf(float f) {
    union { __bf16 h; unsigned short u; } cv;
    cv.h = (__bf16)f;
    return cv.u;
}

__device__ __forceinline__ unsigned int lds_dword(const unsigned short* p) {
    return *reinterpret_cast<const unsigned int*>(p);
}

#if HAVE_ASYNC_LDS
// builtin signature (probe-confirmed): (int4 addrspace(1)*, int4 addrspace(3)*, imm, imm)
typedef int v4i __attribute__((ext_vector_type(4)));
typedef __attribute__((address_space(1))) v4i gv4i;
typedef __attribute__((address_space(3))) v4i lv4i;
#endif

// 16-byte global -> LDS copy; async (ASYNCcnt) when available
__device__ __forceinline__ void cp16_g2l(unsigned short* l, const unsigned short* g) {
#if HAVE_ASYNC_LDS
    __builtin_amdgcn_global_load_async_to_lds_b128((gv4i*)g, (lv4i*)l, 0, 0);
#else
    *reinterpret_cast<uint4*>(l) = *reinterpret_cast<const uint4*>(g);
#endif
}
__device__ __forceinline__ void staging_wait() {
#if HAVE_ASYNC_LDS
    __builtin_amdgcn_s_wait_asynccnt(0);
#endif
}

// ---------------- pre-pass: x f32 -> bf16 row-major + bf16 transposed ----------------
__global__ __launch_bounds__(256)
void convert_bf16(const float* __restrict__ x,
                  unsigned short* __restrict__ xbf,   // [B][L][D]
                  unsigned short* __restrict__ xT) {  // [B][D][L]
    __shared__ unsigned int tile[64 * 65];            // [d][l], 1 bf16/dword, pad->no conflicts
    const int tid = threadIdx.x;
    const int bid = blockIdx.x;
    const int dt  = bid & (D_DIM / 64 - 1);           // 0..3
    const int lt  = (bid >> 2) & (L_SEQ / 64 - 1);    // 0..31
    const int b   = bid >> 7;
    const int l0  = lt * 64, d0 = dt * 64;

    const float*    xs = x   + ((size_t)b * L_SEQ + l0) * D_DIM + d0;
    unsigned short* os = xbf + ((size_t)b * L_SEQ + l0) * D_DIM + d0;
#pragma unroll
    for (int i = 0; i < 8; ++i) {
        const int e   = i * 256 + tid;
        const int row = e >> 5;                       // l 0..63
        const int dp  = (e & 31) * 2;                 // d pair
        const float2 v = *reinterpret_cast<const float2*>(xs + (size_t)row * D_DIM + dp);
        const unsigned short h0 = f2bf(v.x), h1 = f2bf(v.y);
        *reinterpret_cast<unsigned int*>(os + (size_t)row * D_DIM + dp) =
            (unsigned int)h0 | ((unsigned int)h1 << 16);
        tile[(dp + 0) * 65 + row] = h0;
        tile[(dp + 1) * 65 + row] = h1;
    }
    __syncthreads();
    unsigned short* ot = xT + ((size_t)b * D_DIM + d0) * L_SEQ + l0;
#pragma unroll
    for (int i = 0; i < 8; ++i) {
        const int e    = i * 256 + tid;
        const int drow = e >> 5;                      // d 0..63
        const int lp   = (e & 31) * 2;                // l pair
        const unsigned int u0 = tile[drow * 65 + lp];
        const unsigned int u1 = tile[drow * 65 + lp + 1];
        *reinterpret_cast<unsigned int*>(ot + (size_t)drow * L_SEQ + lp) = u0 | (u1 << 16);
    }
}

// ---------------- fused attention ----------------
template <bool PRE>
__global__ __launch_bounds__(256, 1)
void attn_fused(const float* __restrict__ x,
                const unsigned short* __restrict__ xbf,
                const unsigned short* __restrict__ xT,
                const unsigned char* __restrict__ xmask,
                float* __restrict__ out) {
    constexpr int KBUFS = PRE ? 2 : 1;   // double-buffer K/V tiles on the async path
    extern __shared__ __align__(16) unsigned short smem[];
    unsigned short* lds_q = smem;                             // [Q_TILE][D_DIM]
    unsigned short* lds_k = lds_q + Q_TILE * D_DIM;           // KBUFS x [K_TILE][D_DIM]
    unsigned short* lds_v = lds_k + KBUFS * K_TILE * D_DIM;   // KBUFS x [D_DIM][K_TILE]
    unsigned short* lds_p = lds_v + KBUFS * D_DIM * K_TILE;   // [N_WAVE][16][K_TILE]

    const int tid  = threadIdx.x;
    const int wave = tid >> 5;
    const int lane = tid & 31;
    const int lg   = lane >> 4;
    const int ln   = lane & 15;

    const int b     = blockIdx.x / (L_SEQ / Q_TILE);
    const int qbase = (blockIdx.x % (L_SEQ / Q_TILE)) * Q_TILE;

    const float*          xb   = x     + (size_t)b * L_SEQ * D_DIM;
    const unsigned short* xbfb = PRE ? xbf + (size_t)b * L_SEQ * D_DIM : nullptr;
    const unsigned short* xTb  = PRE ? xT  + (size_t)b * D_DIM * L_SEQ : nullptr;
    const unsigned char*  mb   = xmask + (size_t)b * L_SEQ;

    // async-stage one 64-key K/V tile into buffer `buf` (PRE path only)
    auto stage_tile = [&](int kk, int buf) {
        unsigned short* dk = lds_k + buf * (K_TILE * D_DIM);
        unsigned short* dv = lds_v + buf * (D_DIM * K_TILE);
#pragma unroll 1
        for (int i = 0; i < 8; ++i) {       // K: 64x256 bf16 = 2048 16B chunks
            const int e = i * 256 + tid;
            cp16_g2l(dk + (e >> 5) * D_DIM + (e & 31) * 8,
                     xbfb + (size_t)(kk + (e >> 5)) * D_DIM + (e & 31) * 8);
        }
#pragma unroll 1
        for (int i = 0; i < 8; ++i) {       // V^T: 256x64 bf16 = 2048 16B chunks
            const int e = i * 256 + tid;
            cp16_g2l(dv + (e >> 3) * K_TILE + (e & 7) * 8,
                     xTb + (size_t)(e >> 3) * L_SEQ + kk + (e & 7) * 8);
        }
    };

    // ---- stage Q tile (+ prologue tile 0 on the async path) ----
    if constexpr (PRE) {
#pragma unroll 1
        for (int i = 0; i < 16; ++i) {      // Q: 128x256 bf16 = 4096 16B chunks
            const int e   = i * 256 + tid;
            const int row = e >> 5;
            const int c8  = (e & 31) * 8;
            cp16_g2l(lds_q + row * D_DIM + c8, xbfb + (size_t)(qbase + row) * D_DIM + c8);
        }
        stage_tile(0, 0);
        staging_wait();
        __syncthreads();
    } else {
#pragma unroll 1
        for (int i = 0; i < 32; ++i) {
            const int f4  = i * 256 + tid;
            const int row = f4 >> 6;
            const int c4  = (f4 & 63) << 2;
            const float4 v = *reinterpret_cast<const float4*>(xb + (size_t)(qbase + row) * D_DIM + c4);
            const unsigned long long p =
                (unsigned long long)f2bf(v.x)         | ((unsigned long long)f2bf(v.y) << 16) |
                ((unsigned long long)f2bf(v.z) << 32) | ((unsigned long long)f2bf(v.w) << 48);
            *reinterpret_cast<unsigned long long*>(lds_q + row * D_DIM + c4) = p;
        }
    }

    // ---- per-wave flash-attention state ----
    v8f o[16];
    const v8f vzero = {0.f,0.f,0.f,0.f,0.f,0.f,0.f,0.f};
#pragma unroll
    for (int t = 0; t < 16; ++t) o[t] = vzero;

    float rm[8], rs[8];
#pragma unroll
    for (int r = 0; r < 8; ++r) { rm[r] = -3.0e38f; rs[r] = 0.0f; }

    const int qrow_base = qbase + wave * 16;
    const int w4        = wave & 3;
    const int diag_k0   = qrow_base & ~(K_TILE - 1);
    unsigned short* pscr = lds_p + wave * (16 * K_TILE);

#pragma unroll 1
    for (int k0 = 0; k0 < L_SEQ; k0 += K_TILE) {
        const int buf = PRE ? ((k0 / K_TILE) & 1) : 0;
        const unsigned short* dk = lds_k + buf * (K_TILE * D_DIM);
        const unsigned short* dv = lds_v + buf * (D_DIM * K_TILE);

        if constexpr (PRE) {
            // kick off async copy of the NEXT tile into the other buffer,
            // then compute on the current one; wait + barrier at iteration end
            if (k0 + K_TILE < L_SEQ) stage_tile(k0 + K_TILE, buf ^ 1);
            if (k0 + 2 * K_TILE < L_SEQ) {   // prefetch tile k+2 toward L2
                __builtin_prefetch(xbfb + (size_t)(k0 + 2 * K_TILE + (tid >> 2)) * D_DIM + (tid & 3) * 64, 0, 0);
                __builtin_prefetch(xTb + (size_t)tid * L_SEQ + k0 + 2 * K_TILE, 0, 0);
            }
        } else {
            __syncthreads();
#pragma unroll 1
            for (int i = 0; i < 16; ++i) {
                const int f4  = i * 256 + tid;
                const int row = f4 >> 6;
                const int c4  = (f4 & 63) << 2;
                const float4 v = *reinterpret_cast<const float4*>(xb + (size_t)(k0 + row) * D_DIM + c4);
                const unsigned short h0 = f2bf(v.x), h1 = f2bf(v.y), h2 = f2bf(v.z), h3 = f2bf(v.w);
                const unsigned long long p =
                    (unsigned long long)h0 | ((unsigned long long)h1 << 16) |
                    ((unsigned long long)h2 << 32) | ((unsigned long long)h3 << 48);
                *reinterpret_cast<unsigned long long*>(lds_k + row * D_DIM + c4) = p;
                lds_v[(c4 + 0) * K_TILE + row] = h0;
                lds_v[(c4 + 1) * K_TILE + row] = h1;
                lds_v[(c4 + 2) * K_TILE + row] = h2;
                lds_v[(c4 + 3) * K_TILE + row] = h3;
            }
            __syncthreads();
            if (k0 + K_TILE < L_SEQ) {
                const float* nt = xb + (size_t)(k0 + K_TILE + (tid >> 2)) * D_DIM + (tid & 3) * 64;
                __builtin_prefetch(nt, 0, 0);
                __builtin_prefetch(nt + 32, 0, 0);
            }
        }

        // ---- S = Q · K^T : four 16x16 f32 tiles, contracting D in 8 chunks ----
        v8f sAcc[N_SUB];
#pragma unroll
        for (int s = 0; s < N_SUB; ++s) sAcc[s] = vzero;

#pragma unroll
        for (int c = 0; c < D_DIM / 32; ++c) {
            Frag qf;   // A 16x32 bf16: lane=M, half-pairs -> K with 8-interleave
            const unsigned short* qp = lds_q + (wave * 16 + ln) * D_DIM + c * 32 + lg * 8;
            qf.u[0] = lds_dword(qp +  0); qf.u[1] = lds_dword(qp +  2);
            qf.u[2] = lds_dword(qp +  4); qf.u[3] = lds_dword(qp +  6);
            qf.u[4] = lds_dword(qp + 16); qf.u[5] = lds_dword(qp + 18);
            qf.u[6] = lds_dword(qp + 20); qf.u[7] = lds_dword(qp + 22);
#pragma unroll
            for (int s = 0; s < N_SUB; ++s) {
                Frag kf;   // B 32x16: B[kdim=d][n=key] = x[key][d]
                const unsigned short* kp = dk + (s * 16 + ln) * D_DIM + c * 32 + lg * 8;
#pragma unroll
                for (int i = 0; i < 8; ++i) kf.u[i] = lds_dword(kp + 2 * i + (i >> 2) * 8);
                sAcc[s] = __builtin_amdgcn_wmma_f32_16x16x32_bf16(false, qf.v, false, kf.v,
                                                                  (short)0, sAcc[s], false, false);
            }
        }

        // ---- diagonal zeroing: only in the single tile containing q==k ----
        if (k0 == diag_k0) {
#pragma unroll
            for (int s = 0; s < N_SUB; ++s) {
                const bool subhit = (s == w4);
#pragma unroll
                for (int r = 0; r < 8; ++r)
                    if (subhit && (ln == r + lg * 8)) sAcc[s][r] = 0.0f;
            }
        }

        // ---- key-padding mask + online softmax (C layout: M = r + lg*8, N = ln) ----
        bool mk[N_SUB];
#pragma unroll
        for (int s = 0; s < N_SUB; ++s) mk[s] = mb[k0 + s * 16 + ln] != 0;

        float scale[8];
#pragma unroll
        for (int r = 0; r < 8; ++r) {
            float a[N_SUB];
#pragma unroll
            for (int s = 0; s < N_SUB; ++s)
                a[s] = mk[s] ? -3.0e38f : sAcc[s][r];

            float tmax = fmaxf(fmaxf(a[0], a[1]), fmaxf(a[2], a[3]));
#pragma unroll
            for (int off = 1; off < 16; off <<= 1)
                tmax = fmaxf(tmax, __shfl_xor(tmax, off, 32));
            const float nm = fmaxf(rm[r], tmax);
            const float sc = __expf(rm[r] - nm);
            rm[r] = nm;

            float p[N_SUB];
#pragma unroll
            for (int s = 0; s < N_SUB; ++s) p[s] = __expf(a[s] - nm);
            float psum = (p[0] + p[1]) + (p[2] + p[3]);
#pragma unroll
            for (int off = 1; off < 16; off <<= 1)
                psum += __shfl_xor(psum, off, 32);
            rs[r] = rs[r] * sc + psum;
            scale[r] = sc;

            unsigned short* pp = pscr + (r + lg * 8) * K_TILE;
#pragma unroll
            for (int s = 0; s < N_SUB; ++s) pp[s * 16 + ln] = f2bf(p[s]);
        }

        // rescale O accumulators
#pragma unroll
        for (int t = 0; t < 16; ++t)
#pragma unroll
            for (int r = 0; r < 8; ++r)
                o[t][r] *= scale[r];

        // ---- O += P · V : A = P (two 16x32 bf16), B = V (two 32x16 per D-chunk) ----
        Frag pf0, pf1;
        {
            const unsigned short* pp0 = pscr + ln * K_TILE + lg * 8;
            const unsigned short* pp1 = pp0 + 32;
            pf0.u[0] = lds_dword(pp0 +  0); pf0.u[1] = lds_dword(pp0 +  2);
            pf0.u[2] = lds_dword(pp0 +  4); pf0.u[3] = lds_dword(pp0 +  6);
            pf0.u[4] = lds_dword(pp0 + 16); pf0.u[5] = lds_dword(pp0 + 18);
            pf0.u[6] = lds_dword(pp0 + 20); pf0.u[7] = lds_dword(pp0 + 22);
            pf1.u[0] = lds_dword(pp1 +  0); pf1.u[1] = lds_dword(pp1 +  2);
            pf1.u[2] = lds_dword(pp1 +  4); pf1.u[3] = lds_dword(pp1 +  6);
            pf1.u[4] = lds_dword(pp1 + 16); pf1.u[5] = lds_dword(pp1 + 18);
            pf1.u[6] = lds_dword(pp1 + 20); pf1.u[7] = lds_dword(pp1 + 22);
        }
#pragma unroll
        for (int t = 0; t < 16; ++t) {
            Frag vf0, vf1;
            const unsigned short* vp0 = dv + (t * 16 + ln) * K_TILE + lg * 8;
            const unsigned short* vp1 = vp0 + 32;
#pragma unroll
            for (int i = 0; i < 8; ++i) { vf0.u[i] = lds_dword(vp0 + 2 * i + (i >> 2) * 8);
                                          vf1.u[i] = lds_dword(vp1 + 2 * i + (i >> 2) * 8); }
            o[t] = __builtin_amdgcn_wmma_f32_16x16x32_bf16(false, pf0.v, false, vf0.v,
                                                           (short)0, o[t], false, false);
            o[t] = __builtin_amdgcn_wmma_f32_16x16x32_bf16(false, pf1.v, false, vf1.v,
                                                           (short)0, o[t], false, false);
        }

        if constexpr (PRE) {
            staging_wait();     // next-tile async copies complete
            __syncthreads();    // all waves done reading current buffers
        }
    }

    // ---- finalize ----
    float inv[8];
#pragma unroll
    for (int r = 0; r < 8; ++r) inv[r] = 1.0f / rs[r];
#pragma unroll
    for (int t = 0; t < 16; ++t) {
#pragma unroll
        for (int r = 0; r < 8; ++r) {
            const int qg = qrow_base + r + lg * 8;
            out[((size_t)b * L_SEQ + qg) * D_DIM + t * 16 + ln] = o[t][r] * inv[r];
        }
    }
}

extern "C" void kernel_launch(void* const* d_in, const int* in_sizes, int n_in,
                              void* d_out, int out_size, void* d_ws, size_t ws_size,
                              hipStream_t stream) {
    (void)in_sizes; (void)n_in; (void)out_size;
    const float*         x     = (const float*)d_in[0];
    const unsigned char* xmask = (const unsigned char*)d_in[1];  // jax bool = 1 byte
    float*               out   = (float*)d_out;

    dim3 grid(B_SZ * (L_SEQ / Q_TILE));   // 128 workgroups
    dim3 block(256);                      // 8 wave32 waves

    const size_t elems = (size_t)B_SZ * L_SEQ * D_DIM;
    const size_t need  = 2 * elems * sizeof(unsigned short);   // bf16 + bf16-transposed

    if (ws_size >= need) {
        const size_t shmem = (size_t)(Q_TILE * D_DIM + 2 * K_TILE * D_DIM +
                                      2 * D_DIM * K_TILE + N_WAVE * 16 * K_TILE) * sizeof(unsigned short);
        unsigned short* xbf = (unsigned short*)d_ws;
        unsigned short* xT  = xbf + elems;
        hipLaunchKernelGGL(convert_bf16, dim3(B_SZ * (L_SEQ / 64) * (D_DIM / 64)), dim3(256),
                           0, stream, x, xbf, xT);
        hipLaunchKernelGGL((attn_fused<true>), grid, block, shmem, stream,
                           x, xbf, xT, xmask, out);
    } else {
        const size_t shmem = (size_t)(Q_TILE * D_DIM + K_TILE * D_DIM +
                                      D_DIM * K_TILE + N_WAVE * 16 * K_TILE) * sizeof(unsigned short);
        hipLaunchKernelGGL((attn_fused<false>), grid, block, shmem, stream,
                           x, nullptr, nullptr, xmask, out);
    }
}
